// policy_model_83562883711691
// MI455X (gfx1250) — matrix operand order repaired and checked by reading, not compiled, not run on hardware
//
#include <hip/hip_runtime.h>
#include <hip/hip_bf16.h>

typedef __bf16 bf16;
typedef __attribute__((ext_vector_type(16))) __bf16 v16bf;
typedef __attribute__((ext_vector_type(8)))  float  v8f;

#define IM   896
#define MBSZ 10
#define DIMR 30
#define HID  128

static __device__ __forceinline__ float bf2f(bf16 x) { return (float)x; }
static __device__ __forceinline__ bf16  f2bf(float x) { return (bf16)x; }

// ---------------------------------------------------------------- utilities
__global__ void k_zero(float* p, int n) {
    int i = blockIdx.x * 256 + threadIdx.x;
    if (i < n) p[i] = 0.f;
}

// Pack fp32 weight (O rows x Kreal cols, row-major) into per-wave WMMA B fragments:
// out[((ntile*Ktiles + kt)*32 + lane)*16 + i], n = ntile*16 + (lane&15),
// k = kt*32 + ((lane>>4)&1)*8 + (i&7) + ((i>>3)<<4).  Zero-padded for k >= Kreal.
__global__ void k_pack_frag(const float* __restrict__ W, bf16* __restrict__ out,
                            int O, int Kreal, int Ktiles) {
    int idx = blockIdx.x * 256 + threadIdx.x;
    int total = (O >> 4) * Ktiles * 32 * 16;
    if (idx >= total) return;
    int i    = idx & 15;
    int lane = (idx >> 4) & 31;
    int rest = idx >> 9;
    int kt    = rest % Ktiles;
    int ntile = rest / Ktiles;
    int n = ntile * 16 + (lane & 15);
    int k = kt * 32 + ((lane >> 4) & 1) * 8 + (i & 7) + ((i >> 3) << 4);
    out[idx] = (k < Kreal) ? f2bf(W[(long long)n * Kreal + k]) : f2bf(0.f);
}

// Transpose fp32 [384][128] -> [128][384] for coalesced GRU matvec.
__global__ void k_packT(const float* __restrict__ W, float* __restrict__ out) {
    int idx = blockIdx.x * 256 + threadIdx.x;
    if (idx >= 384 * 128) return;
    int r = idx / 128, k = idx % 128;
    out[k * 384 + r] = W[idx];
}

// Per-pixel: smallest batch index b whose box contains the pixel (255 = none).
__global__ void k_minb(const int* __restrict__ sp, const int* __restrict__ itrp,
                       unsigned char* __restrict__ minb) {
    int idx = blockIdx.x * 256 + threadIdx.x;
    if (idx >= IM * IM) return;
    int r = idx / IM, c = idx % IM;
    int itr = itrp[0];
    unsigned char mb = 255;
    for (int b = MBSZ - 1; b >= 0; --b) {
        int l0 = sp[(itr * MBSZ + b) * 2 + 0];
        int l1 = sp[(itr * MBSZ + b) * 2 + 1];
        int r0 = max(l0 - DIMR, 0), r1 = min(l0 + DIMR, IM - 1);
        int c0 = max(l1 - DIMR, 0), c1 = min(l1 + DIMR, IM - 1);
        if (r >= r0 && r < r1 && c >= c0 && c < c1) mb = (unsigned char)b;
    }
    minb[idx] = mb;
}

// ------------------------------------------------ conv1: fused patch-select + stats
// One thread = one (b, oh, ow), all 8 output channels; each input tap loaded once.
// Grid is exact: 10*448*448 / 256 = 7840 blocks, no tail.
__global__ void k_conv1(const float* __restrict__ inp, const float* __restrict__ lastp,
                        const unsigned char* __restrict__ minb,
                        const float* __restrict__ w, const float* __restrict__ bias,
                        bf16* __restrict__ out, float* __restrict__ stats) {
    __shared__ float sw[216];
    __shared__ float red[256];
    const int Ho = 448, Wo = 448;
    int tid = threadIdx.x;
    if (tid < 216) sw[tid] = w[tid];
    __syncthreads();
    int idx = blockIdx.x * 256 + tid;            // over b*Ho*Wo
    int ow = idx % Wo; int t = idx / Wo;
    int oh = t % Ho;   int b = t / Ho;
    const float* last9 = lastp + (size_t)9 * 3 * IM * IM;
    float acc[8];
#pragma unroll
    for (int oc = 0; oc < 8; ++oc) acc[oc] = bias[oc];
#pragma unroll
    for (int ic = 0; ic < 3; ++ic)
#pragma unroll
        for (int ky = 0; ky < 3; ++ky) {
            int ih = oh * 2 - 1 + ky;
            bool ihok = (unsigned)ih < (unsigned)IM;
#pragma unroll
            for (int kx = 0; kx < 3; ++kx) {
                int iw = ow * 2 - 1 + kx;
                float v = 0.f;
                if (ihok && (unsigned)iw < (unsigned)IM) {
                    int pix = ih * IM + iw;
                    const float* src = (minb[pix] <= b) ? inp : last9;
                    v = src[((size_t)ic * IM + ih) * IM + iw];
                }
#pragma unroll
                for (int oc = 0; oc < 8; ++oc)
                    acc[oc] += v * sw[oc * 27 + ic * 9 + ky * 3 + kx];
            }
        }
#pragma unroll
    for (int oc = 0; oc < 8; ++oc)
        out[(((size_t)b * 8 + oc) * Ho + oh) * Wo + ow] = f2bf(acc[oc]);
    // fused BN stats: block tree-reduce per channel, one atomic pair per block/channel
    for (int oc = 0; oc < 8; ++oc) {
        red[tid] = acc[oc];
        __syncthreads();
        for (int s = 128; s > 0; s >>= 1) {
            if (tid < s) red[tid] += red[tid + s];
            __syncthreads();
        }
        float bsum = red[0];
        __syncthreads();
        red[tid] = acc[oc] * acc[oc];
        __syncthreads();
        for (int s = 128; s > 0; s >>= 1) {
            if (tid < s) red[tid] += red[tid + s];
            __syncthreads();
        }
        if (tid == 0) { atomicAdd(&stats[oc], bsum); atomicAdd(&stats[8 + oc], red[0]); }
        __syncthreads();
    }
}

// ---------------------------------------------------------- BN stats (layers 2-4)
__global__ void k_stats(const bf16* __restrict__ x, float* __restrict__ stats,
                        int HW, int C) {
    __shared__ float s1[256], s2[256];
    int plane = blockIdx.y;
    int c = plane % C;
    int i = blockIdx.x * 256 + threadIdx.x;
    float v = 0.f;
    if (i < HW) v = bf2f(x[(long long)plane * HW + i]);
    s1[threadIdx.x] = v; s2[threadIdx.x] = v * v;
    __syncthreads();
    for (int s = 128; s > 0; s >>= 1) {
        if ((int)threadIdx.x < s) {
            s1[threadIdx.x] += s1[threadIdx.x + s];
            s2[threadIdx.x] += s2[threadIdx.x + s];
        }
        __syncthreads();
    }
    if (threadIdx.x == 0) { atomicAdd(&stats[c], s1[0]); atomicAdd(&stats[C + c], s2[0]); }
}

// ---------------------------------------------------------- BN normalize + 2x2 maxpool
__global__ void k_bn_pool(const bf16* __restrict__ x, const float* __restrict__ stats,
                          const float* __restrict__ g, const float* __restrict__ bta,
                          bf16* __restrict__ out, int C, int Hi, int Wi, float invn) {
    int Ho = Hi >> 1, Wo = Wi >> 1;
    long long total = (long long)MBSZ * C * Ho * Wo;
    long long idx = (long long)blockIdx.x * blockDim.x + threadIdx.x;
    if (idx >= total) return;
    int pw = (int)(idx % Wo); long long t = idx / Wo;
    int ph = (int)(t % Ho); t /= Ho;
    int c = (int)(t % C); int b = (int)(t / C);
    float mean = stats[c] * invn;
    float var  = stats[C + c] * invn - mean * mean;
    float sc = g[c] * rsqrtf(var + 1e-5f);
    float sh = bta[c] - mean * sc;
    const bf16* base = x + (((long long)b * C + c) * Hi + 2 * ph) * Wi + 2 * pw;
    float v0 = bf2f(base[0]) * sc + sh,  v1 = bf2f(base[1]) * sc + sh;
    float v2 = bf2f(base[Wi]) * sc + sh, v3 = bf2f(base[Wi + 1]) * sc + sh;
    out[idx] = f2bf(fmaxf(fmaxf(v0, v1), fmaxf(v2, v3)));
}

// ---------------------------------------------------------- implicit-GEMM conv (WMMA bf16)
// Fully-templated shapes: no runtime div/mod, K-tile loop unrolls to back-to-back wmma.
template<int Cin, int Hi, int Wi, int Ho, int Wo, int Cout,
         int STRIDE, int PAD, int KSZ, int Kreal, int Ktiles>
__global__ void k_conv_wmma(const bf16* __restrict__ in, const bf16* __restrict__ Bp,
                            const float* __restrict__ bias, bf16* __restrict__ out) {
    constexpr int Mtot = MBSZ * Ho * Wo;
    constexpr int mtiles = (Mtot + 15) >> 4;
    constexpr int KHW = KSZ * KSZ;
    int lane = threadIdx.x & 31;
    int wid  = blockIdx.x * (blockDim.x >> 5) + (threadIdx.x >> 5);
    if (wid >= mtiles) wid = mtiles - 1;       // wave-uniform clamp: EXEC stays all-ones
    int ntile = blockIdx.y;
    int half = (lane >> 4) & 1;
    int lrow = lane & 15;
    int m = wid * 16 + lrow;
    bool mvalid = m < Mtot;
    int mm = mvalid ? m : 0;
    int ow = mm % Wo; int t = mm / Wo; int oh = t % Ho; int b = t / Ho;
    int n = ntile * 16 + lrow;
    v8f acc = {};
#pragma unroll
    for (int kt = 0; kt < Ktiles; ++kt) {
        v16bf afrag;
        // A: lane row = lane&15, K offsets {half*8 + 0..7} U {half*8 + 16..23}
#pragma unroll
        for (int i = 0; i < 16; ++i) {
            int k = kt * 32 + half * 8 + (i & 7) + ((i >> 3) << 4);
            float v = 0.f;
            if (mvalid && k < Kreal) {
                int ic = k / KHW; int rr = k % KHW; int ky = rr / KSZ; int kx = rr % KSZ;
                int ih = oh * STRIDE - PAD + ky;
                int iw = ow * STRIDE - PAD + kx;
                if ((unsigned)ih < (unsigned)Hi && (unsigned)iw < (unsigned)Wi)
                    v = bf2f(in[(((size_t)b * Cin + ic) * Hi + ih) * Wi + iw]);
            }
            afrag[i] = f2bf(v);
        }
        // B: pre-packed fragment, two contiguous 16B loads per lane
        union { uint4 u[2]; v16bf v; } fb;
        const bf16* bbase = Bp + (((size_t)ntile * Ktiles + kt) * 32 + lane) * 16;
        fb.u[0] = *(const uint4*)(bbase);
        fb.u[1] = *(const uint4*)(bbase + 8);
        acc = __builtin_amdgcn_wmma_f32_16x16x32_bf16(false, afrag, false, fb.v,
                                                      (short)0, acc, false, false);
    }
    float bz = bias[n];
#pragma unroll
    for (int j = 0; j < 8; ++j) {
        int mo = wid * 16 + j + half * 8;  // D: VGPR j lanes0-15 -> M=j, lanes16-31 -> M=j+8
        if (mo < Mtot) {
            int ow2 = mo % Wo; int t2 = mo / Wo; int oh2 = t2 % Ho; int b2 = t2 / Ho;
            out[(((size_t)b2 * Cout + n) * Ho + oh2) * Wo + ow2] = f2bf(acc[j] + bz);
        }
    }
}

// ---------------------------------------------------------- dense GEMM (16 x N x K, WMMA bf16)
// A: bf16 [16][K] (invalid rows pre-zeroed), Bp: fragment-packed, Y: fp32 [Mvalid][N]
template<int K, int N>
__global__ void k_gemm_wmma(const bf16* __restrict__ A, const bf16* __restrict__ Bp,
                            const float* __restrict__ bias, float* __restrict__ Y,
                            int Mvalid) {
    constexpr int Ktiles = K >> 5;
    int lane = threadIdx.x & 31;
    int ntile = blockIdx.x;
    int half = (lane >> 4) & 1;
    int lrow = lane & 15;
    int n = ntile * 16 + lrow;
    v8f acc = {};
#pragma unroll 4
    for (int kt = 0; kt < Ktiles; ++kt) {
        union { uint4 u[2]; v16bf v; } fa;
        const bf16* arow = A + (size_t)lrow * K + kt * 32 + half * 8;
        fa.u[0] = *(const uint4*)(arow);
        fa.u[1] = *(const uint4*)(arow + 16);
        union { uint4 u[2]; v16bf v; } fb;
        const bf16* bbase = Bp + (((size_t)ntile * Ktiles + kt) * 32 + lane) * 16;
        fb.u[0] = *(const uint4*)(bbase);
        fb.u[1] = *(const uint4*)(bbase + 8);
        acc = __builtin_amdgcn_wmma_f32_16x16x32_bf16(false, fa.v, false, fb.v,
                                                      (short)0, acc, false, false);
    }
    float bz = bias[n];
#pragma unroll
    for (int j = 0; j < 8; ++j) {
        int m = j + half * 8;
        if (m < Mvalid) Y[(size_t)m * N + n] = acc[j] + bz;
    }
}

// ---------------------------------------------------------- goal embed + concat -> xcat bf16 [16][256]
__global__ void k_embed_concat(const float* __restrict__ goal, const float* __restrict__ gw,
                               const float* __restrict__ gb, const float* __restrict__ y1,
                               bf16* __restrict__ xcat) {
    int j = threadIdx.x;  // 128 threads
    float acc = gb[j];
    for (int k = 0; k < 40; ++k) acc += goal[k] * gw[j * 40 + k];
    for (int m = 0; m < 16; ++m) {
        float a = (m < MBSZ) ? y1[m * HID + j] : 0.f;
        float g = (m < MBSZ) ? acc : 0.f;
        xcat[m * 256 + j]       = f2bf(a);
        xcat[m * 256 + 128 + j] = f2bf(g);
    }
}

// ---------------------------------------------------------- GRU recurrence (one block per direction)
// whhT: [128][384] transposed weights -> coalesced loads across the 128-thread block.
__global__ void k_gru(const float* __restrict__ gi_all,
                      const float* __restrict__ whhT_f, const float* __restrict__ whhT_b,
                      const float* __restrict__ bhh_f, const float* __restrict__ bhh_b,
                      const float* __restrict__ h0, float* __restrict__ outfb) {
    int dir = blockIdx.x;
    int j = threadIdx.x;  // 128
    const float* gi   = gi_all + (size_t)dir * MBSZ * 384;
    const float* whhT = dir ? whhT_b : whhT_f;
    const float* bhh  = dir ? bhh_b : bhh_f;
    float* out = outfb + (size_t)dir * MBSZ * HID;
    __shared__ float h[HID];
    h[j] = h0[dir * HID + j];
    __syncthreads();
    for (int t = 0; t < MBSZ; ++t) {
        int row = dir ? (MBSZ - 1 - t) : t;
        float ghr = bhh[j], ghz = bhh[128 + j], ghn = bhh[256 + j];
        for (int k = 0; k < HID; ++k) {
            float hk = h[k];
            const float* wrow = whhT + (size_t)k * 384;
            ghr += wrow[j] * hk;
            ghz += wrow[128 + j] * hk;
            ghn += wrow[256 + j] * hk;
        }
        float ir = gi[row * 384 + j], iz = gi[row * 384 + 128 + j], inn = gi[row * 384 + 256 + j];
        float r = 1.f / (1.f + __expf(-(ir + ghr)));
        float z = 1.f / (1.f + __expf(-(iz + ghz)));
        float nn = tanhf(inn + r * ghn);
        float hn = (1.f - z) * nn + z * h[j];
        __syncthreads();
        h[j] = hn;
        __syncthreads();
        out[row * HID + j] = hn;   // backward dir stores already un-reversed
    }
}

// ---------------------------------------------------------- head: fc1 + fc2
__global__ void k_head(const float* __restrict__ outfb,
                       const float* __restrict__ w1, const float* __restrict__ b1,
                       const float* __restrict__ w2, const float* __restrict__ b2,
                       float* __restrict__ out) {
    __shared__ float hid[MBSZ * 64];
    const float* of = outfb;
    const float* ob = outfb + MBSZ * HID;
    for (int i = threadIdx.x; i < MBSZ * 64; i += blockDim.x) {
        int m = i / 64, q = i % 64;
        float acc = b1[q];
        for (int k = 0; k < HID; ++k) acc += of[m * HID + k] * w1[q * 256 + k];
        for (int k = 0; k < HID; ++k) acc += ob[m * HID + k] * w1[q * 256 + 128 + k];
        hid[i] = acc;
    }
    __syncthreads();
    for (int i = threadIdx.x; i < MBSZ * 5; i += blockDim.x) {
        int m = i / 5, q = i % 5;
        float acc = b2[q];
        for (int k = 0; k < 64; ++k) acc += hid[m * 64 + k] * w2[q * 64 + k];
        out[i] = acc;
    }
}

// ================================================================ launch
extern "C" void kernel_launch(void* const* d_in, const int* in_sizes, int n_in,
                              void* d_out, int out_size, void* d_ws, size_t ws_size,
                              hipStream_t stream) {
    const float* inputs  = (const float*)d_in[0];
    const int*   sp      = (const int*)d_in[1];
    const float* goal    = (const float*)d_in[2];
    const float* last    = (const float*)d_in[3];
    const float* h0      = (const float*)d_in[4];
    const int*   itr     = (const int*)d_in[5];
    const float* conv1_w = (const float*)d_in[6],  *conv1_b = (const float*)d_in[7];
    const float* bn1_g   = (const float*)d_in[8],  *bn1_b   = (const float*)d_in[9];
    const float* conv2_w = (const float*)d_in[10], *conv2_b = (const float*)d_in[11];
    const float* bn2_g   = (const float*)d_in[12], *bn2_b   = (const float*)d_in[13];
    const float* conv3_w = (const float*)d_in[14], *conv3_b = (const float*)d_in[15];
    const float* bn3_g   = (const float*)d_in[16], *bn3_b   = (const float*)d_in[17];
    const float* conv4_w = (const float*)d_in[18], *conv4_b = (const float*)d_in[19];
    const float* bn4_g   = (const float*)d_in[20], *bn4_b   = (const float*)d_in[21];
    const float* lin_w   = (const float*)d_in[22], *lin_b   = (const float*)d_in[23];
    const float* goal_w  = (const float*)d_in[24], *goal_b  = (const float*)d_in[25];
    const float* wih_f   = (const float*)d_in[26], *whh_f   = (const float*)d_in[27];
    const float* bih_f   = (const float*)d_in[28], *bhh_f   = (const float*)d_in[29];
    const float* wih_b   = (const float*)d_in[30], *whh_b   = (const float*)d_in[31];
    const float* bih_b   = (const float*)d_in[32], *bhh_b   = (const float*)d_in[33];
    const float* fc1_w   = (const float*)d_in[34], *fc1_b   = (const float*)d_in[35];
    const float* fc2_w   = (const float*)d_in[36], *fc2_b   = (const float*)d_in[37];

    char* ws = (char*)d_ws;
    size_t off = 0;
    auto alloc = [&](size_t bytes) -> char* {
        char* p = ws + off;
        off += (bytes + 255) & ~(size_t)255;
        return p;
    };
    unsigned char* minb = (unsigned char*)alloc((size_t)IM * IM);
    bf16* c1    = (bf16*)alloc((size_t)10 * 8 * 448 * 448 * 2);
    float* stats = (float*)alloc(176 * 4);
    float *st1 = stats, *st2 = stats + 16, *st3 = stats + 48, *st4 = stats + 112;
    bf16* p1    = (bf16*)alloc((size_t)10 * 8 * 224 * 224 * 2);
    bf16* w2p   = (bf16*)alloc((size_t)1 * 3 * 512 * 2);
    bf16* c2    = (bf16*)alloc((size_t)10 * 16 * 112 * 112 * 2);
    bf16* p2    = (bf16*)alloc((size_t)10 * 16 * 56 * 56 * 2);
    bf16* w3p   = (bf16*)alloc((size_t)2 * 5 * 512 * 2);
    bf16* c3    = (bf16*)alloc((size_t)10 * 32 * 28 * 28 * 2);
    bf16* p3    = (bf16*)alloc((size_t)10 * 32 * 14 * 14 * 2);
    bf16* w4p   = (bf16*)alloc((size_t)2 * 1 * 512 * 2);
    bf16* c4    = (bf16*)alloc((size_t)10 * 32 * 16 * 16 * 2);
    bf16* flatA = (bf16*)alloc(16 * 2048 * 2);
    bf16* wlinp = (bf16*)alloc((size_t)8 * 64 * 512 * 2);
    float* y1   = (float*)alloc(10 * 128 * 4);
    bf16* xcat  = (bf16*)alloc(16 * 256 * 2);
    bf16* wihfp = (bf16*)alloc((size_t)24 * 8 * 512 * 2);
    bf16* wihbp = (bf16*)alloc((size_t)24 * 8 * 512 * 2);
    float* whhTf = (float*)alloc((size_t)128 * 384 * 4);
    float* whhTb = (float*)alloc((size_t)128 * 384 * 4);
    float* gi   = (float*)alloc(2 * 10 * 384 * 4);
    float* outfb = (float*)alloc(2 * 10 * 128 * 4);

    // zero stats + flatA (pad rows)
    k_zero<<<1, 256, 0, stream>>>(stats, 176);
    k_zero<<<(16 * 2048 * 2 / 4 + 255) / 256, 256, 0, stream>>>((float*)flatA, 16 * 2048 * 2 / 4);

    // pack weights into WMMA B fragments (bf16) + GRU transpose
    k_pack_frag<<<(1536 + 255) / 256, 256, 0, stream>>>(conv2_w, w2p, 16, 72, 3);
    k_pack_frag<<<(5120 + 255) / 256, 256, 0, stream>>>(conv3_w, w3p, 32, 144, 5);
    k_pack_frag<<<(1024 + 255) / 256, 256, 0, stream>>>(conv4_w, w4p, 32, 32, 1);
    k_pack_frag<<<(262144 + 255) / 256, 256, 0, stream>>>(lin_w, wlinp, 128, 2048, 64);
    k_pack_frag<<<(98304 + 255) / 256, 256, 0, stream>>>(wih_f, wihfp, 384, 256, 8);
    k_pack_frag<<<(98304 + 255) / 256, 256, 0, stream>>>(wih_b, wihbp, 384, 256, 8);
    k_packT<<<(384 * 128 + 255) / 256, 256, 0, stream>>>(whh_f, whhTf);
    k_packT<<<(384 * 128 + 255) / 256, 256, 0, stream>>>(whh_b, whhTb);

    // mask bitmap
    k_minb<<<(IM * IM + 255) / 256, 256, 0, stream>>>(sp, itr, minb);

    // layer 1: fused patch-select conv + stats, then BN + pool
    k_conv1<<<7840, 256, 0, stream>>>(inputs, last, minb, conv1_w, conv1_b, c1, st1);
    {
        long long pt = (long long)10 * 8 * 224 * 224;
        k_bn_pool<<<(unsigned)((pt + 255) / 256), 256, 0, stream>>>(c1, st1, bn1_g, bn1_b,
                                                                    p1, 8, 448, 448, 1.f / 2007040.f);
    }
    // layer 2 (WMMA)
    {
        dim3 g(7840 / 8, 1);
        k_conv_wmma<8, 224, 224, 112, 112, 16, 2, 1, 3, 72, 3>
            <<<g, 256, 0, stream>>>(p1, w2p, conv2_b, c2);
        dim3 gs((12544 + 255) / 256, 160);
        k_stats<<<gs, 256, 0, stream>>>(c2, st2, 12544, 16);
        long long pt = (long long)10 * 16 * 56 * 56;
        k_bn_pool<<<(unsigned)((pt + 255) / 256), 256, 0, stream>>>(c2, st2, bn2_g, bn2_b,
                                                                    p2, 16, 112, 112, 1.f / 125440.f);
    }
    // layer 3 (WMMA)
    {
        dim3 g((490 + 7) / 8, 2);
        k_conv_wmma<16, 56, 56, 28, 28, 32, 2, 1, 3, 144, 5>
            <<<g, 256, 0, stream>>>(p2, w3p, conv3_b, c3);
        dim3 gs((784 + 255) / 256, 320);
        k_stats<<<gs, 256, 0, stream>>>(c3, st3, 784, 32);
        long long pt = (long long)10 * 32 * 14 * 14;
        k_bn_pool<<<(unsigned)((pt + 255) / 256), 256, 0, stream>>>(c3, st3, bn3_g, bn3_b,
                                                                    p3, 32, 28, 28, 1.f / 7840.f);
    }
    // layer 4 (1x1 conv with pad=1, WMMA) -> pooled into flatA rows 0..9
    {
        dim3 g(160 / 8, 2);
        k_conv_wmma<32, 14, 14, 16, 16, 32, 1, 1, 1, 32, 1>
            <<<g, 256, 0, stream>>>(p3, w4p, conv4_b, c4);
        dim3 gs(1, 320);
        k_stats<<<gs, 256, 0, stream>>>(c4, st4, 256, 32);
        long long pt = (long long)10 * 32 * 8 * 8;
        k_bn_pool<<<(unsigned)((pt + 255) / 256), 256, 0, stream>>>(c4, st4, bn4_g, bn4_b,
                                                                    flatA, 32, 16, 16, 1.f / 2560.f);
    }
    // lin_cnn (WMMA GEMM 16x128x2048)
    k_gemm_wmma<2048, 128><<<128 / 16, 32, 0, stream>>>(flatA, wlinp, lin_b, y1, 10);
    // goal embed + concat -> xcat
    k_embed_concat<<<1, 128, 0, stream>>>(goal, goal_w, goal_b, y1, xcat);
    // GRU input GEMMs (WMMA 16x384x256)
    k_gemm_wmma<256, 384><<<384 / 16, 32, 0, stream>>>(xcat, wihfp, bih_f, gi, 10);
    k_gemm_wmma<256, 384><<<384 / 16, 32, 0, stream>>>(xcat, wihbp, bih_b, gi + 10 * 384, 10);
    // GRU recurrence (2 directions)
    k_gru<<<2, 128, 0, stream>>>(gi, whhTf, whhTb, bhh_f, bhh_b, h0, outfb);
    // head
    k_head<<<1, 256, 0, stream>>>(outfb, fc1_w, fc1_b, fc2_w, fc2_b, (float*)d_out);
}
